// DomainSpecificAttention_37563783970897
// MI455X (gfx1250) — matrix operand order repaired and checked by reading, not compiled, not run on hardware
//
#include <hip/hip_runtime.h>

// DomainSpecificAttention for MI455X (gfx1250, wave32, WMMA).
// B=2, S=2048, E=1024, H=16, D=64.
// d_out = [attn_output (B*S*E f32)] ++ [wmask (B*H*S*S f32)].
// Workspace (~42.5 MB): f16 weights, f16 Q/K/Vt, f16 ctx, f32 rowsums.
// k_attn stages K tiles into LDS via global_load_async_to_lds_b128 (ASYNCcnt).

#define B_  2
#define S_  2048
#define E_  1024
#define H_  16
#define D_  64
#define KCH 128   // K rows staged per LDS chunk in k_attn (128*64*2B = 16 KB)

typedef __attribute__((ext_vector_type(16))) _Float16 v16h;
typedef __attribute__((ext_vector_type(8)))  _Float16 v8h;
typedef __attribute__((ext_vector_type(4)))  _Float16 v4h;
typedef __attribute__((ext_vector_type(8)))  float    v8f;
typedef __attribute__((ext_vector_type(4)))  float    v4f;

static __device__ __forceinline__ v8f wmma_f16(v16h a, v16h b, v8f c) {
  // D(f32 16x16) = A(16x32 f16) * B(32x16 f16) + C
  return __builtin_amdgcn_wmma_f32_16x16x32_f16(
      /*neg_a=*/false, a, /*neg_b=*/false, b,
      /*c_mod=*/(short)0, c, /*reuse_a=*/false, /*reuse_b=*/false);
}

// A fragment (16x32) from an f16 row-major matrix, p at (row0, k0), ld elems.
// Lane ln = row M; half 0 holds K [0..7],[16..23], half 1 holds K [8..15],[24..31].
static __device__ __forceinline__ v16h load_a_h(const _Float16* p, int ld,
                                                int ln, int half) {
  const _Float16* r = p + (size_t)ln * ld + half * 8;
  v8h lo = *(const v8h*)r;
  v8h hi = *(const v8h*)(r + 16);
  v16h a;
#pragma unroll
  for (int i = 0; i < 8; ++i) { a[i] = lo[i]; a[8 + i] = hi[i]; }
  return a;
}

// Same A fragment but sourced from f32 with a per-row scale folded in.
static __device__ __forceinline__ v16h load_a_f32(const float* p, int ld,
                                                  int ln, int half, float sc) {
  const float* r = p + (size_t)ln * ld + half * 8;
  v8f lo = *(const v8f*)r;
  v8f hi = *(const v8f*)(r + 16);
  v16h a;
#pragma unroll
  for (int i = 0; i < 8; ++i) {
    a[i]     = (_Float16)(lo[i] * sc);
    a[8 + i] = (_Float16)(hi[i] * sc);
  }
  return a;
}

// B fragment (32x16): column N=ln is row (n0+ln) of a K-contiguous f16 matrix.
// p at (n0, k0). half 0 -> K [0..7],[16..23], half 1 -> K [8..15],[24..31].
static __device__ __forceinline__ v16h load_bt_h(const _Float16* p, int ld,
                                                 int ln, int half) {
  const _Float16* r = p + (size_t)ln * ld + half * 8;
  v8h lo = *(const v8h*)r;
  v8h hi = *(const v8h*)(r + 16);
  v16h b;
#pragma unroll
  for (int i = 0; i < 8; ++i) { b[i] = lo[i]; b[8 + i] = hi[i]; }
  return b;
}

// ---------------- kernel 0: f32 -> f16 weight conversion --------------------
__global__ __launch_bounds__(256) void k_convert(const float* __restrict__ wqkv,
                                                 const float* __restrict__ wout,
                                                 _Float16* __restrict__ wh) {
  size_t i4 = ((size_t)blockIdx.x * blockDim.x + threadIdx.x) * 4;
  const size_t n1 = (size_t)3 * E_ * E_;
  const float* src = (i4 < n1) ? (wqkv + i4) : (wout + (i4 - n1));
  v4f v = *(const v4f*)src;
  v4h o;
#pragma unroll
  for (int i = 0; i < 4; ++i) o[i] = (_Float16)v[i];
  *(v4h*)(wh + i4) = o;
}

// ---------------- kernel 1: QKV projection GEMM -----------------------------
// [B*S,E] x [E,3E] (W stored [3E,E], K-contiguous) -> Qh/Kh [B,H,S,D] f16,
// Vt [B,H,D,S] f16. Q pre-scaled by 1/sqrt(D)=0.125.
__global__ __launch_bounds__(256) void k_proj(const float* __restrict__ q,
                                              const float* __restrict__ k,
                                              const float* __restrict__ v,
                                              const float* __restrict__ bqkv,
                                              const _Float16* __restrict__ wh,
                                              _Float16* __restrict__ Qh,
                                              _Float16* __restrict__ Kh,
                                              _Float16* __restrict__ Vt) {
  int wave = threadIdx.x >> 5, lane = threadIdx.x & 31;
  int ln = lane & 15, half = lane >> 4;
  int t = blockIdx.x * 8 + wave;      // 256 row-tiles * 192 col-tiles
  int colTile = t % 192, rowTile = t / 192;
  int n0 = rowTile * 16, c0 = colTile * 16;
  int sector = c0 >> 10;              // 0=q 1=k 2=v
  const float* X = (sector == 0) ? q : (sector == 1) ? k : v;
  const float* Xr = X + (size_t)n0 * E_;
  const _Float16* Wr = wh + (size_t)c0 * E_;
  v8f acc = {};
#pragma unroll 4
  for (int kt = 0; kt < E_ / 32; ++kt) {
    int k0 = kt * 32;
    v16h a = load_a_f32(Xr + k0, E_, ln, half, 1.0f);
    v16h b = load_bt_h(Wr + k0, E_, ln, half);
    acc = wmma_f16(a, b, acc);
  }
  int col = c0 + ln;
  float bias = bqkv[col];
  int ec = col & (E_ - 1);
  int h = ec >> 6, d = ec & 63;
  if (sector == 2) {
#pragma unroll
    for (int r = 0; r < 8; ++r) {
      int n = n0 + r + half * 8;
      int bb = n >> 11, s = n & (S_ - 1);
      Vt[(((size_t)bb * H_ + h) * D_ + d) * S_ + s] = (_Float16)(acc[r] + bias);
    }
  } else {
    float sc = (sector == 0) ? 0.125f : 1.0f;   // fold 1/sqrt(D) into Q
    _Float16* dst = (sector == 0) ? Qh : Kh;
#pragma unroll
    for (int r = 0; r < 8; ++r) {
      int n = n0 + r + half * 8;
      int bb = n >> 11, s = n & (S_ - 1);
      dst[(((size_t)bb * H_ + h) * S_ + s) * D_ + d] =
          (_Float16)((acc[r] + bias) * sc);
    }
  }
}

// ---------------- kernel 2: scores + softmax (unnormalized exp) -------------
// One wave per (b,h, 16-row q block); all 8 waves of a block share (b,h), so
// K tiles are staged once per block into LDS with async-to-LDS copies:
//   global_load_async_to_lds_b128  (ASYNCcnt) -> s_wait_asynccnt -> barrier.
// Pass 1: row max (lane-local, one shuffle reduce at the end). Pass 2:
// recompute scores from LDS, store exp(s-max) into the wmask region of d_out,
// accumulate row sums -> rowsum[B*H*S]. (Softmax denominator is deferred.)
__global__ __launch_bounds__(256) void k_attn(const _Float16* __restrict__ Qh,
                                              const _Float16* __restrict__ Kh,
                                              float* __restrict__ attn,
                                              float* __restrict__ rowsum) {
  __shared__ _Float16 kbuf[KCH * D_];
  int wave = threadIdx.x >> 5, lane = threadIdx.x & 31;
  int ln = lane & 15, half = lane >> 4;
  int gw = blockIdx.x * 8 + wave;     // B*H * S/16 = 4096 waves
  int bh = gw >> 7, qt = gw & 127;
  const _Float16* Qp = Qh + ((size_t)bh * S_ + qt * 16) * D_;
  const _Float16* Kp = Kh + (size_t)bh * S_ * D_;
  v16h a0 = load_a_h(Qp, D_, ln, half);        // K = 0..31
  v16h a1 = load_a_h(Qp + 32, D_, ln, half);   // K = 32..63

  // Per-thread 64B slice of the 16KB chunk; generic LDS address truncates to
  // the 32-bit LDS byte offset (aperture bits live in addr[63:32]).
  unsigned ldsDst = (unsigned)(uintptr_t)kbuf + (unsigned)threadIdx.x * 64u;

  float m[8];
#pragma unroll
  for (int r = 0; r < 8; ++r) m[r] = -3.4e38f;

  for (int ch = 0; ch < S_ / KCH; ++ch) {
    const char* src =
        (const char*)(Kp + (size_t)ch * KCH * D_) + (size_t)threadIdx.x * 64;
    asm volatile(
        "global_load_async_to_lds_b128 %0, %1, off\n\t"
        "global_load_async_to_lds_b128 %0, %1, off offset:16\n\t"
        "global_load_async_to_lds_b128 %0, %1, off offset:32\n\t"
        "global_load_async_to_lds_b128 %0, %1, off offset:48"
        :: "v"(ldsDst), "v"(src) : "memory");
    asm volatile("s_wait_asynccnt 0x0" ::: "memory");
    __syncthreads();
#pragma unroll
    for (int tt = 0; tt < KCH / 16; ++tt) {
      const _Float16* kp = kbuf + tt * 16 * D_;
      v8f c = {};
      c = wmma_f16(a0, load_bt_h(kp, D_, ln, half), c);
      c = wmma_f16(a1, load_bt_h(kp + 32, D_, ln, half), c);
#pragma unroll
      for (int r = 0; r < 8; ++r) m[r] = fmaxf(m[r], c[r]);
    }
    __syncthreads();
  }
#pragma unroll
  for (int r = 0; r < 8; ++r) {                // reduce over 16 cols/half
    m[r] = fmaxf(m[r], __shfl_xor(m[r], 1));
    m[r] = fmaxf(m[r], __shfl_xor(m[r], 2));
    m[r] = fmaxf(m[r], __shfl_xor(m[r], 4));
    m[r] = fmaxf(m[r], __shfl_xor(m[r], 8));
  }

  float l[8] = {0.f, 0.f, 0.f, 0.f, 0.f, 0.f, 0.f, 0.f};
  float* arow = attn + ((size_t)bh * S_ + qt * 16) * S_;
  for (int ch = 0; ch < S_ / KCH; ++ch) {
    const char* src =
        (const char*)(Kp + (size_t)ch * KCH * D_) + (size_t)threadIdx.x * 64;
    asm volatile(
        "global_load_async_to_lds_b128 %0, %1, off\n\t"
        "global_load_async_to_lds_b128 %0, %1, off offset:16\n\t"
        "global_load_async_to_lds_b128 %0, %1, off offset:32\n\t"
        "global_load_async_to_lds_b128 %0, %1, off offset:48"
        :: "v"(ldsDst), "v"(src) : "memory");
    asm volatile("s_wait_asynccnt 0x0" ::: "memory");
    __syncthreads();
#pragma unroll
    for (int tt = 0; tt < KCH / 16; ++tt) {
      const _Float16* kp = kbuf + tt * 16 * D_;
      v8f c = {};
      c = wmma_f16(a0, load_bt_h(kp, D_, ln, half), c);
      c = wmma_f16(a1, load_bt_h(kp + 32, D_, ln, half), c);
      int ct = ch * (KCH / 16) + tt;
#pragma unroll
      for (int r = 0; r < 8; ++r) {
        float e = __expf(c[r] - m[r]);
        l[r] += e;
        arow[(size_t)(r + half * 8) * S_ + ct * 16 + ln] = e;
      }
    }
    __syncthreads();
  }
#pragma unroll
  for (int r = 0; r < 8; ++r) {
    l[r] += __shfl_xor(l[r], 1);
    l[r] += __shfl_xor(l[r], 2);
    l[r] += __shfl_xor(l[r], 4);
    l[r] += __shfl_xor(l[r], 8);
  }
  if (ln == 0) {
#pragma unroll
    for (int r = 0; r < 8; ++r)
      rowsum[(size_t)bh * S_ + qt * 16 + half * 8 + r] = l[r];
  }
}

// ---------------- kernel 3: ctx = softmax(attn) @ V -------------------------
// 1/rowsum folded into the f32->f16 A conversion. Vt is [B,H,D,S] so B
// fragments are contiguous. Output ctxf [B,S,E] f16 (= [B*S, E] row-major).
__global__ __launch_bounds__(256) void k_ctx(const float* __restrict__ attn,
                                             const float* __restrict__ rowsum,
                                             const _Float16* __restrict__ Vt,
                                             _Float16* __restrict__ ctxf) {
  int wave = threadIdx.x >> 5, lane = threadIdx.x & 31;
  int ln = lane & 15, half = lane >> 4;
  int gw = blockIdx.x * 8 + wave;
  int bh = gw >> 7, qt = gw & 127;
  float invl = 1.0f / rowsum[(size_t)bh * S_ + qt * 16 + ln];  // A row = ln
  const float* Ap = attn + ((size_t)bh * S_ + qt * 16) * S_;
  const _Float16* Vp = Vt + (size_t)bh * D_ * S_;
  v8f c0 = {}, c1 = {}, c2 = {}, c3 = {};
  for (int kt = 0; kt < S_ / 32; ++kt) {
    int k0 = kt * 32;
    v16h a = load_a_f32(Ap + k0, S_, ln, half, invl);
    c0 = wmma_f16(a, load_bt_h(Vp + (size_t)0 * 16 * S_ + k0, S_, ln, half), c0);
    c1 = wmma_f16(a, load_bt_h(Vp + (size_t)1 * 16 * S_ + k0, S_, ln, half), c1);
    c2 = wmma_f16(a, load_bt_h(Vp + (size_t)2 * 16 * S_ + k0, S_, ln, half), c2);
    c3 = wmma_f16(a, load_bt_h(Vp + (size_t)3 * 16 * S_ + k0, S_, ln, half), c3);
  }
  int bb = bh >> 4, h = bh & 15;
  _Float16* crow = ctxf + ((size_t)bb * S_ + qt * 16) * E_ + h * D_;
#pragma unroll
  for (int r = 0; r < 8; ++r) {
    _Float16* o = crow + (size_t)(r + half * 8) * E_;
    o[0 * 16 + ln] = (_Float16)c0[r];
    o[1 * 16 + ln] = (_Float16)c1[r];
    o[2 * 16 + ln] = (_Float16)c2[r];
    o[3 * 16 + ln] = (_Float16)c3[r];
  }
}

// ---------------- kernel 4: output projection + bias ------------------------
__global__ __launch_bounds__(256) void k_outproj(const _Float16* __restrict__ ctxf,
                                                 const _Float16* __restrict__ wout_h,
                                                 const float* __restrict__ bout,
                                                 float* __restrict__ out) {
  int wave = threadIdx.x >> 5, lane = threadIdx.x & 31;
  int ln = lane & 15, half = lane >> 4;
  int t = blockIdx.x * 8 + wave;      // 256 row-tiles * 64 col-tiles
  int colTile = t & 63, rowTile = t >> 6;
  int n0 = rowTile * 16, c0 = colTile * 16;
  const _Float16* Ar = ctxf + (size_t)n0 * E_;
  const _Float16* Wr = wout_h + (size_t)c0 * E_;
  v8f acc = {};
#pragma unroll 4
  for (int kt = 0; kt < E_ / 32; ++kt) {
    int k0 = kt * 32;
    acc = wmma_f16(load_a_h(Ar + k0, E_, ln, half),
                   load_bt_h(Wr + k0, E_, ln, half), acc);
  }
  float bias = bout[c0 + ln];
#pragma unroll
  for (int r = 0; r < 8; ++r)
    out[(size_t)(n0 + r + half * 8) * E_ + c0 + ln] = acc[r] + bias;
}

// ---------------- kernel 5: in-place masked renormalization -----------------
// wmask[row,:] = e*mask / sum(e*mask); the softmax denominator cancels, so the
// unnormalized exp stored by k_attn is exactly what we need here.
__global__ __launch_bounds__(256) void k_wmask(float* __restrict__ attn,
                                               const float* __restrict__ dmask) {
  int row = blockIdx.x;               // B*H*S rows
  float* w = attn + (size_t)row * S_;
  int bb = row >> 15;                 // row / (H*S)
  const float* mk = dmask + bb * S_;
  float p = 0.f;
  for (int j = threadIdx.x; j < S_; j += 256) p += w[j] * mk[j];
#pragma unroll
  for (int msk = 1; msk < 32; msk <<= 1) p += __shfl_xor(p, msk);
  __shared__ float red[8];
  int wave = threadIdx.x >> 5, lane = threadIdx.x & 31;
  if (lane == 0) red[wave] = p;
  __syncthreads();
  if (wave == 0) {
    float tsum = (lane < 8) ? red[lane] : 0.f;
    tsum += __shfl_xor(tsum, 1);
    tsum += __shfl_xor(tsum, 2);
    tsum += __shfl_xor(tsum, 4);
    if (lane == 0) red[0] = tsum;
  }
  __syncthreads();
  float inv = 1.0f / red[0];
  for (int j = threadIdx.x; j < S_; j += 256) w[j] = w[j] * mk[j] * inv;
}

extern "C" void kernel_launch(void* const* d_in, const int* in_sizes, int n_in,
                              void* d_out, int out_size, void* d_ws, size_t ws_size,
                              hipStream_t stream) {
  (void)in_sizes; (void)n_in; (void)out_size; (void)ws_size;
  const float* q     = (const float*)d_in[0];
  const float* k     = (const float*)d_in[1];
  const float* v     = (const float*)d_in[2];
  const float* dmask = (const float*)d_in[3];
  const float* wqkv  = (const float*)d_in[4];
  const float* bqkv  = (const float*)d_in[5];
  const float* wout  = (const float*)d_in[6];
  const float* bout  = (const float*)d_in[7];

  char* ws = (char*)d_ws;
  size_t off = 0;
  auto carve = [&](size_t bytes) {
    char* p = ws + off;
    off = (off + bytes + 255) & ~(size_t)255;
    return p;
  };
  _Float16* Wh   = (_Float16*)carve((size_t)4 * E_ * E_ * sizeof(_Float16)); // qkv+out
  _Float16* Qh   = (_Float16*)carve((size_t)B_ * H_ * S_ * D_ * sizeof(_Float16));
  _Float16* Kh   = (_Float16*)carve((size_t)B_ * H_ * S_ * D_ * sizeof(_Float16));
  _Float16* Vt   = (_Float16*)carve((size_t)B_ * H_ * S_ * D_ * sizeof(_Float16));
  _Float16* ctxf = (_Float16*)carve((size_t)B_ * S_ * E_ * sizeof(_Float16));
  float*    rsum = (float*)   carve((size_t)B_ * H_ * S_ * sizeof(float));

  float* out  = (float*)d_out;
  float* attn = out + (size_t)B_ * S_ * E_;          // wmask region

  k_convert<<<4096, 256, 0, stream>>>(wqkv, wout, Wh);
  k_proj   <<<6144, 256, 0, stream>>>(q, k, v, bqkv, Wh, Qh, Kh, Vt);
  k_attn   <<<512,  256, 0, stream>>>(Qh, Kh, attn, rsum);
  k_ctx    <<<512,  256, 0, stream>>>(attn, rsum, Vt, ctxf);
  k_outproj<<<2048, 256, 0, stream>>>(ctxf, Wh + (size_t)3 * E_ * E_, bout, out);
  k_wmask  <<<B_ * H_ * S_, 256, 0, stream>>>(attn, dmask);
}
